// SurfaceFeaturePropagationCD_81870666596693
// MI455X (gfx1250) — matrix operand order repaired and checked by reading, not compiled, not run on hardware
//
#include <hip/hip_runtime.h>
#include <hip/hip_bf16.h>

typedef __attribute__((ext_vector_type(16))) _Float16 v16h;
typedef __attribute__((ext_vector_type(8)))  _Float16 v8h;
typedef __attribute__((ext_vector_type(4)))  _Float16 v4h;
typedef __attribute__((ext_vector_type(8)))  float    v8f;

#define B_    8
#define N1_   8192
#define N2_   2048
#define C1_   128
#define C2_   256
#define D0_   256
#define D1_   128

#define LDA_  40   // padded LDS row stride (halfwords): 80B, 16B-aligned, conflict-free

// ---------------------------------------------------------------------------
// Y[M,N] = A[M,K] @ W[K,N] + bias   (f32 in/out, f16 WMMA compute, f32 accum)
// ATRANS==1: A stored channel-major per batch ([B,K,NPTS]): elem(r=b*NPTS+n,k)
//            at A[b*K*NPTS + k*NPTS + n]
// Block = 8 waves; tile = 128 rows x 32 cols (2 wmma tiles per wave).
// All strides compile-time; global loads are b128; fragments are ds_load_b128.
// ---------------------------------------------------------------------------
template<int K, int N, int ATRANS, int NPTS>
__global__ __launch_bounds__(256)
void gemm_wmma(const float* __restrict__ A, const float* __restrict__ W,
               const float* __restrict__ bias, float* __restrict__ Y, int M)
{
    __shared__ _Float16 sA[128 * LDA_];   // 10 KB, layout [row][k]
    __shared__ _Float16 sB[32 * LDA_];    // 2.5 KB, layout [n][k] (transposed!)

    const int t  = threadIdx.x;
    const int l  = t & 31;
    const int w  = t >> 5;
    const int hl = l >> 4;
    const int lm = l & 15;
    const int rowBase = blockIdx.x * 128;
    const int n0 = blockIdx.y * 32;

    v8f acc0 = {};
    v8f acc1 = {};

    for (int kk = 0; kk < K; kk += 32) {
        // ================= stage A tile (128 rows x 32 k) =================
        if constexpr (ATRANS) {
            const int b  = rowBase / NPTS;
            const int nb = rowBase % NPTS;
            const float* Ab = A + (size_t)b * K * NPTS + (size_t)kk * NPTS + nb;
            if (kk + 32 < K)
                __builtin_prefetch(Ab + (size_t)32 * NPTS + ((t & 31) * 4), 0, 3);
            float4 va[4];
            #pragma unroll
            for (int i = 0; i < 4; ++i) {               // issue 4x global_load_b128
                int e = t + i * 256;
                int kidx = e >> 5;                      // 0..31
                int r4   = (e & 31) * 4;                // 0..124
                va[i] = *(const float4*)&Ab[(size_t)kidx * NPTS + r4];
            }
            #pragma unroll
            for (int i = 0; i < 4; ++i) {               // convert + scatter to LDS
                int e = t + i * 256;
                int kidx = e >> 5;
                int r4   = (e & 31) * 4;
                sA[(r4 + 0) * LDA_ + kidx] = (_Float16)va[i].x;
                sA[(r4 + 1) * LDA_ + kidx] = (_Float16)va[i].y;
                sA[(r4 + 2) * LDA_ + kidx] = (_Float16)va[i].z;
                sA[(r4 + 3) * LDA_ + kidx] = (_Float16)va[i].w;
            }
        } else {
            const float* Ar = A + (size_t)rowBase * K + kk;
            if (kk + 32 < K)
                __builtin_prefetch(Ar + 32 + (size_t)(t >> 3) * K, 0, 3);
            float4 va[4];
            #pragma unroll
            for (int i = 0; i < 4; ++i) {               // issue 4x global_load_b128
                int e = t + i * 256;
                int r  = e >> 3;
                int k4 = (e & 7) * 4;
                va[i] = *(const float4*)&Ar[(size_t)r * K + k4];
            }
            #pragma unroll
            for (int i = 0; i < 4; ++i) {               // packed ds_store_b64
                int e = t + i * 256;
                int r  = e >> 3;
                int k4 = (e & 7) * 4;
                v4h pk;
                pk[0] = (_Float16)va[i].x; pk[1] = (_Float16)va[i].y;
                pk[2] = (_Float16)va[i].z; pk[3] = (_Float16)va[i].w;
                *(v4h*)&sA[r * LDA_ + k4] = pk;
            }
        }
        // ====== stage B tile transposed: sB[n][k], one b128 per thread ======
        {
            const int k  = t >> 3;                      // 0..31
            const int n4 = (t & 7) * 4;                 // 0..28, contiguous in W row
            float4 vb = *(const float4*)&W[(size_t)(kk + k) * N + n0 + n4];
            sB[(n4 + 0) * LDA_ + k] = (_Float16)vb.x;
            sB[(n4 + 1) * LDA_ + k] = (_Float16)vb.y;
            sB[(n4 + 2) * LDA_ + k] = (_Float16)vb.z;
            sB[(n4 + 3) * LDA_ + k] = (_Float16)vb.w;
        }
        __syncthreads();

        // ============ fragments: 2x ds_load_b128 per operand ============
        const int mrow = w * 16 + lm;
        v8h a0 = *(const v8h*)&sA[mrow * LDA_ + hl * 8];        // K = hl*8..+7
        v8h a1 = *(const v8h*)&sA[mrow * LDA_ + 16 + hl * 8];   // K = 16+hl*8..+7
        v16h af = __builtin_shufflevector(a0, a1,
            0,1,2,3,4,5,6,7,8,9,10,11,12,13,14,15);

        v8h b0lo = *(const v8h*)&sB[lm * LDA_ + hl * 8];        // col lm,  K=hl*8.., 16+hl*8..
        v8h b0hi = *(const v8h*)&sB[lm * LDA_ + 16 + hl * 8];
        v16h bf0 = __builtin_shufflevector(b0lo, b0hi,
            0,1,2,3,4,5,6,7,8,9,10,11,12,13,14,15);

        v8h b1lo = *(const v8h*)&sB[(16 + lm) * LDA_ + hl * 8]; // col 16+lm
        v8h b1hi = *(const v8h*)&sB[(16 + lm) * LDA_ + 16 + hl * 8];
        v16h bf1 = __builtin_shufflevector(b1lo, b1hi,
            0,1,2,3,4,5,6,7,8,9,10,11,12,13,14,15);

        acc0 = __builtin_amdgcn_wmma_f32_16x16x32_f16(false, af, false, bf0,
                                                      (short)0, acc0, false, false);
        acc1 = __builtin_amdgcn_wmma_f32_16x16x32_f16(false, af, false, bf1,
                                                      (short)0, acc1, false, false);
        __syncthreads();
    }

    // ---- epilogue: +bias, store f32 (C/D layout: elem j -> M = j + hl*8)
    const int ncol = n0 + lm;
    const float bv0 = bias[ncol];
    const float bv1 = bias[ncol + 16];
    #pragma unroll
    for (int j = 0; j < 8; ++j) {
        int m = rowBase + w * 16 + hl * 8 + j;
        Y[(size_t)m * N + ncol]      = acc0[j] + bv0;
        Y[(size_t)m * N + ncol + 16] = acc1[j] + bv1;
    }
    (void)M;
}

// ---------------------------------------------------------------------------
// Per-channel sum / sum-of-squares over rows (blockDim.x == N channels)
// ---------------------------------------------------------------------------
__global__ void colstats_kernel(const float* __restrict__ Y, int M, int N,
                                float* __restrict__ sums)
{
    const int c   = threadIdx.x;
    const int rpb = M / gridDim.x;
    const int r0  = blockIdx.x * rpb;
    float s = 0.f, sq = 0.f;
    for (int r = 0; r < rpb; ++r) {
        float v = Y[(size_t)(r0 + r) * N + c];   // coalesced across threads
        s += v; sq += v * v;
    }
    atomicAdd(&sums[c], s);
    atomicAdd(&sums[N + c], sq);
}

__global__ void bnfinalize_kernel(const float* __restrict__ sums,
                                  const float* __restrict__ g, const float* __restrict__ be,
                                  int M, int N, float* __restrict__ scale,
                                  float* __restrict__ shift)
{
    int c = threadIdx.x;
    if (c < N) {
        float mean = sums[c] / (float)M;
        float var  = sums[N + c] / (float)M - mean * mean;   // biased, matches ref
        float rs   = rsqrtf(var + 1e-5f);
        float sc   = g[c] * rs;
        scale[c] = sc;
        shift[c] = be[c] - mean * sc;
    }
}

__global__ void zero_kernel(float* __restrict__ p, int n) {
    for (int i = threadIdx.x + blockIdx.x * blockDim.x; i < n; i += blockDim.x * gridDim.x)
        p[i] = 0.f;
}

// ---------------------------------------------------------------------------
// three_nn: per query (thread) brute-force top-3 over 2048 LDS-resident points
// ---------------------------------------------------------------------------
__global__ __launch_bounds__(256)
void three_nn_kernel(const float* __restrict__ center1, const float* __restrict__ center2,
                     int* __restrict__ idx, float* __restrict__ wgt)
{
    __shared__ float sx[N2_], sy[N2_], sz[N2_];   // 24 KB
    const int b = blockIdx.y;
    const int t = threadIdx.x;
    const int n = blockIdx.x * 256 + t;

    const float* c2b = center2 + (size_t)b * 3 * N2_;
    for (int j = t; j < N2_; j += 256) {
        sx[j] = c2b[j];
        sy[j] = c2b[N2_ + j];
        sz[j] = c2b[2 * N2_ + j];
    }
    __syncthreads();

    const float* c1b = center1 + (size_t)b * 3 * N1_;
    const float qx = c1b[n], qy = c1b[N1_ + n], qz = c1b[2 * N1_ + n];

    float d0 = 1e30f, d1 = 1e30f, d2 = 1e30f;
    int   i0 = 0, i1 = 0, i2 = 0;
    for (int j = 0; j < N2_; ++j) {          // LDS broadcast reads
        float dx = qx - sx[j], dy = qy - sy[j], dz = qz - sz[j];
        float d = dx * dx + dy * dy + dz * dz;
        if (d < d0)      { d2 = d1; i2 = i1; d1 = d0; i1 = i0; d0 = d; i0 = j; }
        else if (d < d1) { d2 = d1; i2 = i1; d1 = d;  i1 = j; }
        else if (d < d2) { d2 = d;  i2 = j; }
    }
    float r0 = 1.f / (d0 + 1e-8f), r1 = 1.f / (d1 + 1e-8f), r2 = 1.f / (d2 + 1e-8f);
    float rs = r0 + r1 + r2;
    size_t row = (size_t)b * N1_ + n;
    idx[row * 3 + 0] = i0; idx[row * 3 + 1] = i1; idx[row * 3 + 2] = i2;
    wgt[row * 3 + 0] = r0 / rs; wgt[row * 3 + 1] = r1 / rs; wgt[row * 3 + 2] = r2 / rs;
}

// ---------------------------------------------------------------------------
// x = relu( BN(interp of y2) + BN(skip) );  BN folded to scale/shift, Σw == 1
// ---------------------------------------------------------------------------
__global__ __launch_bounds__(256)
void interp_kernel(const float* __restrict__ y2, const float* __restrict__ skipraw,
                   const int* __restrict__ idx, const float* __restrict__ wgt,
                   const float* __restrict__ scf0, const float* __restrict__ shf0,
                   const float* __restrict__ scs0, const float* __restrict__ shs0,
                   float* __restrict__ xout)
{
    const size_t row = blockIdx.x;
    const int c = threadIdx.x;
    const int b = (int)(row >> 13);                 // row / N1_
    const int i0 = idx[row * 3 + 0], i1 = idx[row * 3 + 1], i2 = idx[row * 3 + 2];
    const float w0 = wgt[row * 3 + 0], w1 = wgt[row * 3 + 1], w2 = wgt[row * 3 + 2];
    const float* yb = y2 + (size_t)b * N2_ * D0_;
    float g = w0 * yb[(size_t)i0 * D0_ + c]
            + w1 * yb[(size_t)i1 * D0_ + c]
            + w2 * yb[(size_t)i2 * D0_ + c];
    g = g * scf0[c] + shf0[c];
    float s = skipraw[row * D0_ + c] * scs0[c] + shs0[c];
    xout[row * D0_ + c] = fmaxf(g + s, 0.f);
}

// ---------------------------------------------------------------------------
// out[b, c, n] = relu( BN(z[b*N1+n, c]) )  -> [B, D1, N1]
// ---------------------------------------------------------------------------
__global__ __launch_bounds__(256)
void final_kernel(const float* __restrict__ z, const float* __restrict__ scc1,
                  const float* __restrict__ shc1, float* __restrict__ out)
{
    const int b = blockIdx.x;
    const int n = blockIdx.y * 256 + threadIdx.x;
    const float* zr = z + ((size_t)b * N1_ + n) * D1_;   // per-thread contiguous row
    float* ob = out + (size_t)b * D1_ * N1_;
    #pragma unroll 4
    for (int c = 0; c < D1_; ++c) {
        float v = zr[c] * scc1[c] + shc1[c];
        ob[(size_t)c * N1_ + n] = fmaxf(v, 0.f);         // coalesced writes
    }
}

extern "C" void kernel_launch(void* const* d_in, const int* in_sizes, int n_in,
                              void* d_out, int out_size, void* d_ws, size_t ws_size,
                              hipStream_t stream)
{
    const float* center1 = (const float*)d_in[0];
    const float* feat1   = (const float*)d_in[1];
    const float* center2 = (const float*)d_in[2];
    const float* feat2   = (const float*)d_in[3];
    const float* W_f0  = (const float*)d_in[4];
    const float* b_f0  = (const float*)d_in[5];
    const float* g_f0  = (const float*)d_in[6];
    const float* be_f0 = (const float*)d_in[7];
    const float* W_s0  = (const float*)d_in[8];
    const float* b_s0  = (const float*)d_in[9];
    const float* g_s0  = (const float*)d_in[10];
    const float* be_s0 = (const float*)d_in[11];
    const float* W_c1  = (const float*)d_in[12];
    const float* b_c1  = (const float*)d_in[13];
    const float* g_c1  = (const float*)d_in[14];
    const float* be_c1 = (const float*)d_in[15];
    (void)in_sizes; (void)n_in; (void)out_size; (void)ws_size;

    float* ws    = (float*)d_ws;
    float* y2raw = ws;                                   // 16384*256
    float* skipx = y2raw + (size_t)16384 * 256;          // 65536*256 (skip raw -> x)
    float* zbuf  = skipx + (size_t)65536 * 256;          // 65536*128
    int*   idxb  = (int*)(zbuf + (size_t)65536 * 128);   // 65536*3
    float* wgtb  = (float*)(idxb + (size_t)65536 * 3);   // 65536*3
    float* stats = wgtb + (size_t)65536 * 3;
    float* sums_f0  = stats;            // 512
    float* scale_f0 = stats + 512;      // 256
    float* shift_f0 = stats + 768;      // 256
    float* sums_s0  = stats + 1024;     // 512
    float* scale_s0 = stats + 1536;
    float* shift_s0 = stats + 1792;
    float* sums_c1  = stats + 2048;     // 256
    float* scale_c1 = stats + 2304;     // 128
    float* shift_c1 = stats + 2432;     // 128

    zero_kernel<<<4, 256, 0, stream>>>(stats, 2560);

    three_nn_kernel<<<dim3(N1_ / 256, B_), 256, 0, stream>>>(center1, center2, idxb, wgtb);

    // f0: [B*N2, C2] @ [C2, D0]
    gemm_wmma<C2_, D0_, 1, N2_><<<dim3((B_ * N2_) / 128, D0_ / 32), 256, 0, stream>>>(
        feat2, W_f0, b_f0, y2raw, B_ * N2_);
    colstats_kernel<<<64, D0_, 0, stream>>>(y2raw, B_ * N2_, D0_, sums_f0);
    bnfinalize_kernel<<<1, D0_, 0, stream>>>(sums_f0, g_f0, be_f0, B_ * N2_, D0_,
                                             scale_f0, shift_f0);

    // s0: [B*N1, C1] @ [C1, D0]
    gemm_wmma<C1_, D0_, 1, N1_><<<dim3((B_ * N1_) / 128, D0_ / 32), 256, 0, stream>>>(
        feat1, W_s0, b_s0, skipx, B_ * N1_);
    colstats_kernel<<<64, D0_, 0, stream>>>(skipx, B_ * N1_, D0_, sums_s0);
    bnfinalize_kernel<<<1, D0_, 0, stream>>>(sums_s0, g_s0, be_s0, B_ * N1_, D0_,
                                             scale_s0, shift_s0);

    // x = relu(interp + skip), in place over skip buffer
    interp_kernel<<<B_ * N1_, D0_, 0, stream>>>(y2raw, skipx, idxb, wgtb,
                                                scale_f0, shift_f0, scale_s0, shift_s0,
                                                skipx);

    // c1: [B*N1, D0] @ [D0, D1]
    gemm_wmma<D0_, D1_, 0, 1><<<dim3((B_ * N1_) / 128, D1_ / 32), 256, 0, stream>>>(
        skipx, W_c1, b_c1, zbuf, B_ * N1_);
    colstats_kernel<<<64, D1_, 0, stream>>>(zbuf, B_ * N1_, D1_, sums_c1);
    bnfinalize_kernel<<<1, D1_, 0, stream>>>(sums_c1, g_c1, be_c1, B_ * N1_, D1_,
                                             scale_c1, shift_c1);

    final_kernel<<<dim3(B_, N1_ / 256), 256, 0, stream>>>(zbuf, scale_c1, shift_c1,
                                                          (float*)d_out);
}
